// LSTM_16638703305445
// MI455X (gfx1250) — 8-bit matrix operand order repaired and checked by reading, compiled, not run on hardware
//
#include <hip/hip_runtime.h>
#include <hip/hip_bf16.h>
#include <stdint.h>


typedef __attribute__((ext_vector_type(16))) int   v16i;
typedef __attribute__((ext_vector_type(8)))  int   v8i;
typedef __attribute__((ext_vector_type(8)))  float v8f;

#define T_STEPS  365
#define IN_DIM   32
#define H_DIM    256
#define BTILE    16
#define NTHREADS 512
#define SROW     320

#if defined(__has_builtin)
#if __has_builtin(__builtin_amdgcn_cvt_pk_fp8_f32)
#define HAVE_HW_FP8 1
#endif
#endif

__device__ __forceinline__ unsigned f32_to_fp8_sw(float x) {
  unsigned u = __float_as_uint(x);
  unsigned s = (u >> 31) << 7;
  int ef = (int)((u >> 23) & 0xFF);
  if (ef == 0xFF) return s | 0x7E;
  int e = ef - 127;
  unsigned man = u & 0x7FFFFFu;
  if (ef == 0 || e < -9) return s;
  unsigned m = (man + 0x80000u) >> 20;
  if (m >= 8) { m = 0; e += 1; }
  if (e > 8) return s | 0x7E;
  if (e < -6) return s;
  return s | ((unsigned)(e + 7) << 3) | m;
}

__device__ __forceinline__ unsigned pack_fp8x4(float a, float b, float c, float d) {
#ifdef HAVE_HW_FP8
  int v = 0;
  v = __builtin_amdgcn_cvt_pk_fp8_f32(a, b, v, false);
  v = __builtin_amdgcn_cvt_pk_fp8_f32(c, d, v, true);
  return (unsigned)v;
#else
  return f32_to_fp8_sw(a) | (f32_to_fp8_sw(b) << 8) |
         (f32_to_fp8_sw(c) << 16) | (f32_to_fp8_sw(d) << 24);
#endif
}

__device__ __forceinline__ unsigned char fp8_1(float x) {
#ifdef HAVE_HW_FP8
  return (unsigned char)(__builtin_amdgcn_cvt_pk_fp8_f32(x, x, 0, false) & 0xFF);
#else
  return (unsigned char)f32_to_fp8_sw(x);
#endif
}

#define LOG2E  1.4426950408889634f

__device__ __forceinline__ float fast_sigmoid(float x) {
  float e = __builtin_amdgcn_exp2f(-LOG2E * x);
  return __builtin_amdgcn_rcpf(1.0f + e);
}
__device__ __forceinline__ float fast_tanh(float x) {
  float e = __builtin_amdgcn_exp2f(-2.0f * LOG2E * x);
  return (1.0f - e) * __builtin_amdgcn_rcpf(1.0f + e);
}

__global__ __launch_bounds__(NTHREADS)
void lstm_persistent(const float* __restrict__ xd,
                     const float* __restrict__ w_ih,
                     const float* __restrict__ w_hh,
                     const float* __restrict__ bvec,
                     const float* __restrict__ w_out,
                     const float* __restrict__ b_out,
                     float* __restrict__ out)
{
  __shared__ __align__(16) unsigned char lds_whh[64 * 2 * 2048];
  __shared__ __align__(16) unsigned char lds_wih[64 * 512];
  __shared__ __align__(16) unsigned char lds_stage[BTILE * SROW];
  __shared__ float lds_oacc[BTILE];

  const int tid   = threadIdx.x;
  const int lane  = tid & 31;
  const int w     = tid >> 5;
  const int wg    = blockIdx.x;
  const int khalf = lane >> 4;
  const int arow  = lane & 15;

  for (int idx = tid; idx < 64 * 2 * 512; idx += NTHREADS) {
    int frag  = idx >> 9;
    int rem   = idx & 511;
    int flane = rem >> 4;
    int v     = rem & 15;
    int n = frag >> 1, kstep = frag & 1;
    int j  = n * 16 + (flane & 15);
    int kk = kstep * 128 + ((v >> 2) << 5) + ((flane >> 4) << 4) + ((v & 3) << 2);
    const float* src = w_hh + (size_t)j * H_DIM + kk;
    *(unsigned*)(lds_whh + frag * 2048 + flane * 64 + v * 4) =
        pack_fp8x4(src[0], src[1], src[2], src[3]);
  }
  for (int idx = tid; idx < 64 * 128; idx += NTHREADS) {
    int n     = idx >> 7;
    int rem   = idx & 127;
    int flane = rem >> 2;
    int v     = rem & 3;
    int j  = n * 16 + (flane & 15);
    int kk = ((flane >> 4) << 4) + (v << 2);
    const float* src = w_ih + (size_t)j * IN_DIM + kk;
    *(unsigned*)(lds_wih + n * 512 + flane * 16 + v * 4) =
        pack_fp8x4(src[0], src[1], src[2], src[3]);
  }
  for (int idx = tid; idx < BTILE * SROW / 4; idx += NTHREADS)
    ((unsigned*)lds_stage)[idx] = 0u;
  if (tid < BTILE) lds_oacc[tid] = 0.0f;
  __syncthreads();

  const int   col    = w * 16 + (lane & 15);
  const float bias_i = bvec[0 * H_DIM + col];
  const float bias_f = bvec[1 * H_DIM + col];
  const float bias_g = bvec[2 * H_DIM + col];
  const float bias_o = bvec[3 * H_DIM + col];
  const float wout_l = w_out[col];

  float c[8], h[8];
#pragma unroll
  for (int r = 0; r < 8; ++r) { c[r] = 0.0f; h[r] = 0.0f; }

  const float* x_base = xd + (size_t)(wg * BTILE) * T_STEPS * IN_DIM;

  for (int t = 0; t < T_STEPS; ++t) {
    if (tid < 128) {
      int r = tid >> 3, ch = tid & 7;
      const float* xp = x_base + ((size_t)r * T_STEPS + t) * IN_DIM + ch * 4;
      float4 xv = *(const float4*)xp;
      *(unsigned*)(lds_stage + r * SROW + 256 + ch * 4) =
          pack_fp8x4(xv.x, xv.y, xv.z, xv.w);
      if (t + 1 < T_STEPS) __builtin_prefetch(xp + IN_DIM, 0, 1);
    }
    __syncthreads();

    v8f acc[4];
#pragma unroll
    for (int g = 0; g < 4; ++g)
#pragma unroll
      for (int r = 0; r < 8; ++r) acc[g][r] = 0.0f;

#pragma unroll
    for (int kstep = 0; kstep < 2; ++kstep) {
      v16i A;
#pragma unroll
      for (int v = 0; v < 16; ++v) {
        int vv = v & 7;
        int kb = ((v >> 2) << 5) + (khalf << 4) + ((v & 3) << 2);
        A[v] = *(const int*)(lds_stage + arow * SROW + (kstep << 7) + kb);
      }
#pragma unroll
      for (int g = 0; g < 4; ++g) {
        int n = g * 16 + w;
        const int4* q = (const int4*)(lds_whh + (n * 2 + kstep) * 2048 + lane * 64);
        int4 q0 = q[0], q1 = q[1], q2 = q[2], q3 = q[3];
        v16i B;
        B[0]  = q0.x; B[1]  = q0.y; B[2]  = q0.z; B[3]  = q0.w;
        B[4]  = q1.x; B[5]  = q1.y; B[6]  = q1.z; B[7]  = q1.w;
        B[8]  = q2.x; B[9]  = q2.y; B[10] = q2.z; B[11] = q2.w;
        B[12] = q3.x; B[13] = q3.y; B[14] = q3.z; B[15] = q3.w;
        acc[g] = __builtin_amdgcn_wmma_f32_16x16x128_fp8_fp8(
            A, B, (short)0, acc[g], false, false);
      }
    }
    {
      v8i At;
#pragma unroll
      for (int v = 0; v < 8; ++v) {
        int kb = ((v >> 2) << 5) + (khalf << 4) + ((v & 3) << 2);
        At[v] = *(const int*)(lds_stage + arow * SROW + 256 + kb);
      }
#pragma unroll
      for (int g = 0; g < 4; ++g) {
        int n = g * 16 + w;
        const int4 q = *(const int4*)(lds_wih + n * 512 + lane * 16);
        v8i Bt;
        Bt[0] = q.x; Bt[1] = q.y; Bt[2] = q.z; Bt[3] = q.w;
        Bt[4] = 0;   Bt[5] = 0;   Bt[6] = 0;   Bt[7] = 0;
        acc[g] = __builtin_amdgcn_wmma_f32_16x16x64_fp8_fp8(
            At, Bt, (short)0, acc[g], false, false);
      }
    }

#pragma unroll
    for (int r = 0; r < 8; ++r) {
      float gi = fast_sigmoid(acc[0][r] + bias_i);
      float gf = fast_sigmoid(acc[1][r] + bias_f);
      float gg = fast_tanh   (acc[2][r] + bias_g);
      float go = fast_sigmoid(acc[3][r] + bias_o);
      float cc = gf * c[r] + gi * gg;
      c[r] = cc;
      h[r] = go * fast_tanh(cc);
    }
    __syncthreads();

    {
      int hrow = khalf << 3;
#pragma unroll
      for (int r = 0; r < 8; ++r)
        lds_stage[(hrow + r) * SROW + col] = fp8_1(h[r]);
    }
  }

  {
    int hrow = khalf << 3;
#pragma unroll
    for (int r = 0; r < 8; ++r)
      atomicAdd(&lds_oacc[hrow + r], h[r] * wout_l);
  }
  __syncthreads();
  if (tid < BTILE) {
    float v = lds_oacc[tid] + b_out[0];
    out[wg * BTILE + tid] = v > 0.0f ? v : 0.0f;
  }
}

extern "C" void kernel_launch(void* const* d_in, const int* in_sizes, int n_in,
                              void* d_out, int out_size, void* d_ws, size_t ws_size,
                              hipStream_t stream) {
  (void)in_sizes; (void)n_in; (void)out_size; (void)d_ws; (void)ws_size;
  const float* xd    = (const float*)d_in[0];
  const float* w_ih  = (const float*)d_in[1];
  const float* w_hh  = (const float*)d_in[2];
  const float* b     = (const float*)d_in[3];
  const float* w_out = (const float*)d_in[4];
  const float* b_out = (const float*)d_in[5];

  dim3 grid(4096 / BTILE);
  dim3 block(NTHREADS);
  lstm_persistent<<<grid, block, 0, stream>>>(xd, w_ih, w_hh, b, w_out, b_out,
                                              (float*)d_out);
}
